// GraphNN_68676527063643
// MI455X (gfx1250) — matrix operand order repaired and checked by reading, not compiled, not run on hardware
//
#include <hip/hip_runtime.h>

// ---------------------------------------------------------------------------
// Equivariant GNN forward for MI455X (gfx1250, wave32, WMMA bf16).
// Dense GEMMs: v_wmma_f32_16x16x32_bf16, 16x64 tile per wave (4 accumulators,
// A fragment reused across B subtiles; subtile config is a template parameter
// so the hot loop is branch-free). Fused edge kernel keeps the E x nW
// radial-weight tensor out of HBM (WMMA -> LDS -> CG message -> scatter).
// ---------------------------------------------------------------------------

typedef __attribute__((ext_vector_type(16))) __bf16 v16bf;
typedef __attribute__((ext_vector_type(8)))  float  v8f;
typedef __attribute__((ext_vector_type(4)))  float  v4f;

#define CUTOFF   5.0f
#define HID0     32
#define NGATES   12

// CG flat-table offsets (region lives in workspace, filled by k_init_cg)
#define CG000 0
#define CG110 1
#define CG220 10
#define CG011 35
#define CG101 44
#define CG121 53
#define CG211 98
#define CG022 143
#define CG202 168
#define CG112 193
#define CG222 238
#define CG_SZ 363

// Path metadata. Layer 0: [(0,0,0),(0,1,1),(0,2,2)], mul=128.
__constant__ int P0_L1[3]   = {0,0,0};
__constant__ int P0_L2[3]   = {0,1,2};
__constant__ int P0_L3[3]   = {0,1,2};
__constant__ int P0_MUL[3]  = {128,128,128};
__constant__ int P0_WOFF[3] = {0,128,256};
__constant__ int P0_CHOFF[3]= {0,0,0};
__constant__ int P0_CGOFF[3]= {CG000,CG011,CG022};
// Layers 1-3: 11 paths, PATHS_FULL order.
__constant__ int PF_L1[11]   = {0,1,2,0,1,1,2,0,2,1,2};
__constant__ int PF_L2[11]   = {0,1,2,1,0,2,1,2,0,1,2};
__constant__ int PF_L3[11]   = {0,0,0,1,1,1,1,2,2,2,2};
__constant__ int PF_MUL[11]  = {32,8,4,32,8,8,4,32,4,8,4};
__constant__ int PF_WOFF[11] = {0,32,40,44,76,84,92,96,128,132,140};
__constant__ int PF_CHOFF[11]= {0,32,40,0,32,40,48,0,32,36,44};
__constant__ int PF_CGOFF[11]= {CG000,CG110,CG220,CG011,CG101,CG121,CG211,CG022,CG202,CG112,CG222};

__device__ __forceinline__ float siluf(float x){ return x / (1.f + __expf(-x)); }
__device__ __forceinline__ float sigmf(float x){ return 1.f / (1.f + __expf(-x)); }

// Pack 8 consecutive floats (two v4f) into bf16 fragment slots [base..base+7].
__device__ __forceinline__ void pack8(v16bf& d, int base, v4f lo, v4f hi)
{
  d[base+0] = (__bf16)lo[0]; d[base+1] = (__bf16)lo[1];
  d[base+2] = (__bf16)lo[2]; d[base+3] = (__bf16)lo[3];
  d[base+4] = (__bf16)hi[0]; d[base+5] = (__bf16)hi[1];
  d[base+6] = (__bf16)hi[2]; d[base+7] = (__bf16)hi[3];
}

// ---------------------------------------------------------------------------
// FAST GEMM: C = act(alpha * A @ B (+C)), unit column strides, M % 16 == 0.
// One wave computes a 16x(16*NT) tile, NT = NSUB full subtiles (+1 partial of
// NREM cols if NREM>0). All subtile guards are compile-time -> the k-loop is
// pure b128 loads + packed converts + NSUB back-to-back WMMAs, no branches.
// K tail handled by unconditional clamped loads + VALU masking (A/B buffers
// have slack past the tail rows).
// ---------------------------------------------------------------------------
template<int NSUB, int NREM>
__global__ void k_gemm_fast(const float* __restrict__ A, const float* __restrict__ B,
                            float* __restrict__ C, int M, int K,
                            int ars, int brs, int crs,
                            float alpha, int act, int acc)
{
  constexpr int NT = NSUB + (NREM > 0 ? 1 : 0);
  int lane = threadIdx.x & 31;
  int half = lane >> 4, r = lane & 15;
  long tm = blockIdx.x, tn = blockIdx.y;
  long row = tm * 16 + r;                       // M % 16 == 0: always valid
  const float* arow = A + (size_t)row * ars;
  int n0 = (int)tn * 64;

  v8f av[NT];
  #pragma unroll
  for (int j = 0; j < NT; ++j) {
    v8f z = {0.f,0.f,0.f,0.f,0.f,0.f,0.f,0.f};
    av[j] = z;
  }

  int kfull = K & ~31;
  for (int k0 = 0; k0 < kfull; k0 += 32) {
    v16bf af;
    const float* pa = arow + k0 + half * 8;
    pack8(af, 0, *(const v4f*)(pa),      *(const v4f*)(pa + 4));
    pack8(af, 8, *(const v4f*)(pa + 16), *(const v4f*)(pa + 20));
    __builtin_prefetch(pa + 32, 0, 0);          // speculative, unconditional
    const float* bbase = B + (size_t)(k0 + half * 8) * brs + n0 + r;
    #pragma unroll
    for (int j = 0; j < NSUB; ++j) {
      const float* brow = bbase + j * 16;
      v16bf bfm;
      #pragma unroll
      for (int e = 0; e < 8; ++e) {
        bfm[e]     = (__bf16)brow[(size_t)e * brs];
        bfm[8 + e] = (__bf16)brow[(size_t)(16 + e) * brs];
      }
      av[j] = __builtin_amdgcn_wmma_f32_16x16x32_bf16(false, af, false, bfm,
                                                      (short)0, av[j], false, false);
    }
    if (NREM > 0) {                             // compile-time
      const float* brow = bbase + NSUB * 16;
      v16bf bfm;
      #pragma unroll
      for (int e = 0; e < 8; ++e) {
        bfm[e]     = (__bf16)((r < NREM) ? brow[(size_t)e * brs] : 0.f);
        bfm[8 + e] = (__bf16)((r < NREM) ? brow[(size_t)(16 + e) * brs] : 0.f);
      }
      av[NSUB] = __builtin_amdgcn_wmma_f32_16x16x32_bf16(false, af, false, bfm,
                                                         (short)0, av[NSUB], false, false);
    }
  }
  if (kfull < K) {                              // uniform K-tail block
    v16bf af;
    const float* pa = arow + kfull + half * 8;
    v4f aA = *(const v4f*)(pa),      aB = *(const v4f*)(pa + 4);
    v4f aC = *(const v4f*)(pa + 16), aD = *(const v4f*)(pa + 20);
    #pragma unroll
    for (int e = 0; e < 8; ++e) {
      int k  = kfull + half * 8 + e;
      int k2 = k + 16;
      float va = (e < 4) ? aA[e & 3] : aB[e & 3];
      float vb = (e < 4) ? aC[e & 3] : aD[e & 3];
      af[e]     = (__bf16)((k  < K) ? va : 0.f);
      af[8 + e] = (__bf16)((k2 < K) ? vb : 0.f);
    }
    const float* bbase = B + n0 + r;
    #pragma unroll
    for (int j = 0; j < NSUB; ++j) {
      const float* brow = bbase + j * 16;
      v16bf bfm;
      #pragma unroll
      for (int e = 0; e < 8; ++e) {
        int k  = kfull + half * 8 + e;
        int k2 = k + 16;
        float v  = brow[(size_t)((k  < K) ? k  : 0) * brs];
        float v2 = brow[(size_t)((k2 < K) ? k2 : 0) * brs];
        bfm[e]     = (__bf16)((k  < K) ? v  : 0.f);
        bfm[8 + e] = (__bf16)((k2 < K) ? v2 : 0.f);
      }
      av[j] = __builtin_amdgcn_wmma_f32_16x16x32_bf16(false, af, false, bfm,
                                                      (short)0, av[j], false, false);
    }
    if (NREM > 0) {
      const float* brow = bbase + NSUB * 16;
      v16bf bfm;
      #pragma unroll
      for (int e = 0; e < 8; ++e) {
        int k  = kfull + half * 8 + e;
        int k2 = k + 16;
        bfm[e]     = (__bf16)((r < NREM && k  < K) ? brow[(size_t)k  * brs] : 0.f);
        bfm[8 + e] = (__bf16)((r < NREM && k2 < K) ? brow[(size_t)k2 * brs] : 0.f);
      }
      av[NSUB] = __builtin_amdgcn_wmma_f32_16x16x32_bf16(false, af, false, bfm,
                                                         (short)0, av[NSUB], false, false);
    }
  }
  // Epilogue: full subtiles store unconditionally; partial subtile guarded.
  #pragma unroll
  for (int j = 0; j < NSUB; ++j) {
    int n = n0 + j * 16 + r;
    #pragma unroll
    for (int i = 0; i < 8; ++i) {
      long m = tm * 16 + half * 8 + i;
      size_t idx = (size_t)m * crs + n;
      float v = alpha * av[j][i];
      if (acc) v += C[idx];
      if (act) v = siluf(v);
      C[idx] = v;
    }
  }
  if (NREM > 0) {
    int n = n0 + NSUB * 16 + r;
    if (r < NREM) {
      #pragma unroll
      for (int i = 0; i < 8; ++i) {
        long m = tm * 16 + half * 8 + i;
        size_t idx = (size_t)m * crs + n;
        float v = alpha * av[NSUB][i];
        if (acc) v += C[idx];
        if (act) v = siluf(v);
        C[idx] = v;
      }
    }
  }
}

// ---------------------------------------------------------------------------
// GENERIC strided GEMM (small component matmuls: N=8/4, col-strides 3/5).
// Requires M % 16 == 0 (all call sites use M = N_NODE). All loads are
// UNCONDITIONAL from clamped indices + VALU masking -> no per-element
// exec-mask divergence, loads can be claused with a single wait.
// ---------------------------------------------------------------------------
__global__ void k_gemm_gen(const float* __restrict__ A, const float* __restrict__ B,
                           float* __restrict__ C, int M, int Nn, int K,
                           int ars, int acs, int brs, int bcs, int crs, int ccs,
                           float alpha, int act, int acc)
{
  int lane = threadIdx.x & 31;
  int half = lane >> 4, r = lane & 15;
  long tm = blockIdx.x, tn = blockIdx.y;
  long row = tm * 16 + r;                       // M % 16 == 0: always valid
  const float* arow = A + (size_t)row * ars;
  v8f accv = {0.f,0.f,0.f,0.f,0.f,0.f,0.f,0.f};

  for (int k0 = 0; k0 < K; k0 += 32) {
    v16bf af, bfm;
    #pragma unroll
    for (int e = 0; e < 8; ++e) {
      int k  = k0 + half * 8 + e;
      int k2 = k + 16;
      int kc  = (k  < K) ? k  : 0;              // clamped: always in-bounds
      int kc2 = (k2 < K) ? k2 : 0;
      float va = arow[kc  * acs];
      float vb = arow[kc2 * acs];
      af[e]     = (__bf16)((k  < K) ? va : 0.f);
      af[8 + e] = (__bf16)((k2 < K) ? vb : 0.f);
    }
    int n  = (int)tn * 16 + r;
    int nc = (n < Nn) ? n : 0;
    const float* bcol = B + (size_t)nc * bcs;
    #pragma unroll
    for (int e = 0; e < 8; ++e) {
      int k  = k0 + half * 8 + e;
      int k2 = k + 16;
      float v  = bcol[(size_t)((k  < K) ? k  : 0) * brs];
      float v2 = bcol[(size_t)((k2 < K) ? k2 : 0) * brs];
      bfm[e]     = (__bf16)((k  < K && n < Nn) ? v  : 0.f);
      bfm[8 + e] = (__bf16)((k2 < K && n < Nn) ? v2 : 0.f);
    }
    accv = __builtin_amdgcn_wmma_f32_16x16x32_bf16(false, af, false, bfm,
                                                   (short)0, accv, false, false);
  }
  #pragma unroll
  for (int i = 0; i < 8; ++i) {
    long m = tm * 16 + half * 8 + i;
    int  n = (int)tn * 16 + r;
    if (n < Nn) {
      size_t idx = (size_t)m * crs + (size_t)n * ccs;
      float v = alpha * accv[i];
      if (acc) v += C[idx];
      if (act) v = siluf(v);
      C[idx] = v;
    }
  }
}

// ---------------------------------------------------------------------------
// Edge geometry: spherical harmonics (l=0,1,2; 9 values) + radial basis (8).
// ---------------------------------------------------------------------------
__global__ void k_edge_geom(const float* __restrict__ Rij,
                            float* __restrict__ sh, float* __restrict__ rad, int E)
{
  long e = (long)blockIdx.x * blockDim.x + threadIdx.x;
  if (e >= E) return;
  float x = Rij[e*3+0] / CUTOFF, y = Rij[e*3+1] / CUTOFF, z = Rij[e*3+2] / CUTOFF;
  float len = sqrtf(x*x + y*y + z*z);
  float inv = (len > 0.f) ? (1.f / len) : 1.f;
  float ux = x*inv, uy = y*inv, uz = z*inv;
  const float s3 = 1.7320508f, s5 = 2.2360680f, s15 = 3.8729833f;
  float* sp = sh + e*9;
  sp[0] = 1.f;
  sp[1] = s3*uy; sp[2] = s3*uz; sp[3] = s3*ux;
  sp[4] = s15*ux*uy; sp[5] = s15*uy*uz; sp[6] = 0.5f*s5*(3.f*uz*uz - 1.f);
  sp[7] = s15*ux*uz; sp[8] = 0.5f*s15*(ux*ux - uy*uy);
  float* rp = rad + e*8;
  if (len > 0.f && len < 1.f) {
    float r6 = len*len*len; r6 = r6*r6;
    float r7 = r6*len, r8 = r7*len;
    float env = 1.f - 28.f*r6 + 48.f*r7 - 21.f*r8;
    float c = 1.41421356f / len * env;
    #pragma unroll
    for (int k = 0; k < 8; ++k)
      rp[k] = c * __sinf((float)(k+1) * 3.14159265f * len);
  } else {
    #pragma unroll
    for (int k = 0; k < 8; ++k) rp[k] = 0.f;
  }
}

__global__ void k_embed(const float* __restrict__ embed, const int* __restrict__ species,
                        float* __restrict__ F0, int N)
{
  long t = (long)blockIdx.x * blockDim.x + threadIdx.x;
  if (t >= (long)N * 128) return;
  int n = (int)(t >> 7), u = (int)(t & 127);
  F0[t] = embed[species[n] * 128 + u];
}

// ---------------------------------------------------------------------------
// On-device CG table generation (deterministic; runs each launch).
// ---------------------------------------------------------------------------
__global__ void k_init_cg(float* cg)
{
  if (threadIdx.x != 0 || blockIdx.x != 0) return;
  for (int i = 0; i < CG_SZ; ++i) cg[i] = 0.f;
  cg[CG000] = 1.f;
  for (int a = 0; a < 3; ++a) cg[CG110 + (a*3 + a)] = 0.57735027f;
  for (int a = 0; a < 5; ++a) cg[CG220 + (a*5 + a)] = 0.44721360f;
  for (int b = 0; b < 3; ++b) cg[CG011 + b*3 + b] = 1.f;
  for (int a = 0; a < 3; ++a) cg[CG101 + a*3 + a] = 1.f;
  for (int b = 0; b < 5; ++b) cg[CG022 + b*5 + b] = 1.f;
  for (int a = 0; a < 5; ++a) cg[CG202 + a*5 + a] = 1.f;
  const float K1 = 0.5477226f, K2 = 0.3162278f;
  #define I112(a,b,c) (CG112 + ((a)*3 + (b))*5 + (c))
  cg[I112(2,0,0)] = K1;  cg[I112(0,2,0)] = K1;
  cg[I112(0,1,1)] = K1;  cg[I112(1,0,1)] = K1;
  cg[I112(1,1,2)] = 2.f*K2; cg[I112(0,0,2)] = -K2; cg[I112(2,2,2)] = -K2;
  cg[I112(2,1,3)] = K1;  cg[I112(1,2,3)] = K1;
  cg[I112(2,2,4)] = K1;  cg[I112(0,0,4)] = -K1;
  for (int a = 0; a < 3; ++a)
    for (int b = 0; b < 5; ++b)
      for (int c = 0; c < 3; ++c)
        cg[CG121 + (a*5 + b)*3 + c] = cg[CG112 + (a*3 + c)*5 + b];
  for (int a = 0; a < 5; ++a)
    for (int b = 0; b < 3; ++b)
      for (int c = 0; c < 3; ++c)
        cg[CG211 + (a*3 + b)*3 + c] = cg[CG112 + (b*3 + c)*5 + a];
  for (int a = 0; a < 5; ++a)
    for (int b = 0; b < 5; ++b)
      for (int c = 0; c < 5; ++c)
        cg[CG222 + (a*5 + b)*5 + c] =
            (((a + b + c) & 1) == 0) ? 0.1f * (float)(((a + 2*b + 3*c) % 5) - 2) : 0.f;
}

// ---------------------------------------------------------------------------
// Fused edge kernel: per wave, 16 edges.
//  1) WMMA: W-chunk = h2(16x64,bf16) @ w3(64x16,bf16) -> 16x16 f32
//  2) chunk -> LDS
//  3) lanes form messages (CG x X[sender] x sh x W) and atomic-scatter to agg.
// nW is a multiple of 16 (384 / 144) so every column tile is full -> all
// operand loads are unconditional b128 loads. The A fragment is loaded once
// and amortized over all nW/16 column tiles.
// ---------------------------------------------------------------------------
__global__ void k_edge_msg(const float* __restrict__ h2, const float* __restrict__ w3,
                           const float* __restrict__ sh,
                           const float* __restrict__ x0, const float* __restrict__ x1,
                           const float* __restrict__ x2,
                           const int* __restrict__ snd, const int* __restrict__ rcv,
                           const float* __restrict__ cg,
                           float* agg0, float* agg1, float* agg2,
                           int E, int nW, int first,
                           int m0, int m1, int m2,
                           int ch0, int ch1, int ch2)
{
  __shared__ float Wlds[8][256];
  int lane = threadIdx.x & 31;
  int wave = threadIdx.x >> 5;
  int half = lane >> 4, r = lane & 15;
  long ebase = ((long)blockIdx.x * 8 + wave) * 16;

  // A fragments for K=64 (two k-steps), vectorized from contiguous h2 rows.
  v16bf a0, a1;
  long arow = ebase + r;
  if (ebase + 16 <= E) {                        // uniform fast path
    const float* hrow = h2 + (size_t)arow * 64 + half * 8;
    v4f p0 = *(const v4f*)(hrow);      v4f p1 = *(const v4f*)(hrow + 4);
    v4f p2 = *(const v4f*)(hrow + 16); v4f p3 = *(const v4f*)(hrow + 20);
    v4f p4 = *(const v4f*)(hrow + 32); v4f p5 = *(const v4f*)(hrow + 36);
    v4f p6 = *(const v4f*)(hrow + 48); v4f p7 = *(const v4f*)(hrow + 52);
    pack8(a0, 0, p0, p1); pack8(a0, 8, p2, p3);
    pack8(a1, 0, p4, p5); pack8(a1, 8, p6, p7);
  } else {
    bool rok = arow < E;
    const float* hrow = h2 + (size_t)(rok ? arow : 0) * 64;
    #pragma unroll
    for (int e = 0; e < 8; ++e) {
      int k = half * 8 + e;
      a0[e]     = (__bf16)(rok ? hrow[k]      : 0.f);
      a0[8 + e] = (__bf16)(rok ? hrow[k + 16] : 0.f);
      a1[e]     = (__bf16)(rok ? hrow[k + 32] : 0.f);
      a1[8 + e] = (__bf16)(rok ? hrow[k + 48] : 0.f);
    }
  }

  int njt = nW >> 4;                            // nW % 16 == 0
  for (int jt = 0; jt < njt; ++jt) {
    int j0 = jt * 16;
    v16bf b0, b1;
    {
      const float* brow = w3 + (size_t)(half * 8) * nW + j0 + r;
      #pragma unroll
      for (int e = 0; e < 8; ++e) {
        b0[e]     = (__bf16)brow[(size_t)e * nW];
        b0[8 + e] = (__bf16)brow[(size_t)(16 + e) * nW];
        b1[e]     = (__bf16)brow[(size_t)(32 + e) * nW];
        b1[8 + e] = (__bf16)brow[(size_t)(48 + e) * nW];
      }
    }
    v8f acc = {0.f,0.f,0.f,0.f,0.f,0.f,0.f,0.f};
    acc = __builtin_amdgcn_wmma_f32_16x16x32_bf16(false, a0, false, b0, (short)0, acc, false, false);
    acc = __builtin_amdgcn_wmma_f32_16x16x32_bf16(false, a1, false, b1, (short)0, acc, false, false);

    #pragma unroll
    for (int i = 0; i < 8; ++i)
      Wlds[wave][(half * 8 + i) * 16 + r] = acc[i];
    __syncthreads();

    // 16 edges x 16 cols = 256 (edge,col) pairs, 8 per lane.
    for (int t = 0; t < 8; ++t) {
      int pair = lane * 8 + t;
      int le = pair & 15, lc = pair >> 4;
      long e = ebase + le;
      int g = j0 + lc;
      if (e < E) {
        float w = Wlds[wave][le * 16 + lc];
        int np = first ? 3 : 11;
        int p = -1, u = 0;
        for (int q = 0; q < np; ++q) {
          int wo = first ? P0_WOFF[q] : PF_WOFF[q];
          int mu = first ? P0_MUL[q]  : PF_MUL[q];
          if (g >= wo && g < wo + mu) { p = q; u = g - wo; break; }
        }
        if (p >= 0) {
          int l1 = first ? P0_L1[p] : PF_L1[p];
          int l2 = first ? P0_L2[p] : PF_L2[p];
          int l3 = first ? P0_L3[p] : PF_L3[p];
          int co = first ? P0_CHOFF[p] : PF_CHOFF[p];
          int cgo = first ? P0_CGOFF[p] : PF_CGOFF[p];
          int d1 = 2*l1 + 1, d2 = 2*l2 + 1, d3 = 2*l3 + 1;
          int shoff = (l2 == 0) ? 0 : (l2 == 1) ? 1 : 4;
          long s  = snd[e];
          long rv = rcv[e];
          const float* shp = sh + e * 9 + shoff;
          float* aggp = (l3 == 0) ? agg0 : (l3 == 1) ? agg1 : agg2;
          int chT = (l3 == 0) ? ch0 : (l3 == 1) ? ch1 : ch2;
          for (int c = 0; c < d3; ++c) {
            float m = 0.f;
            for (int a = 0; a < d1; ++a) {
              float xa = (l1 == 0) ? x0[s * m0 + u]
                       : (l1 == 1) ? x1[(s * m1 + u) * 3 + a]
                                   : x2[(s * m2 + u) * 5 + a];
              for (int b = 0; b < d2; ++b) {
                float cgv = cg[cgo + (a * d2 + b) * d3 + c];
                m += cgv * xa * shp[b];
              }
            }
            m *= w;
            atomicAdd(&aggp[(rv * chT + co + u) * d3 + c], m);
          }
        }
      }
    }
    __syncthreads();
  }
}

// ---------------------------------------------------------------------------
// Skip connection: Out[n][v][c] = sum_u f[n][u][c] * skipW[species[n]][u][v]
// ---------------------------------------------------------------------------
__global__ void k_skip(const float* __restrict__ f, const float* __restrict__ sw,
                       const int* __restrict__ species, float* __restrict__ Out,
                       int N, int U, int V, int C)
{
  int n = blockIdx.x;
  int t = threadIdx.x;
  if (n >= N || t >= V * C) return;
  int v = t / C, c = t % C;
  int sp = species[n];
  float a = 0.f;
  for (int u = 0; u < U; ++u)
    a += f[((long)n * U + u) * C + c] * sw[((long)sp * U + u) * V + v];
  Out[((long)n * V + v) * C + c] = a;
}

// ---------------------------------------------------------------------------
// Gating: feats0' = silu(s[:32]); gates = sigmoid(s[32:44]);
//         feats1' = out1 * gates[:8]; feats2' = out2 * gates[8:12].
// ---------------------------------------------------------------------------
__global__ void k_gate(const float* __restrict__ O0, const float* __restrict__ O1,
                       const float* __restrict__ O2,
                       float* __restrict__ F0, float* __restrict__ F1,
                       float* __restrict__ F2, int N)
{
  long n = (long)blockIdx.x * blockDim.x + threadIdx.x;
  if (n >= N) return;
  const float* s = O0 + n * 44;
  #pragma unroll
  for (int v = 0; v < HID0; ++v) F0[n * 32 + v] = siluf(s[v]);
  float g[NGATES];
  #pragma unroll
  for (int j = 0; j < NGATES; ++j) g[j] = sigmf(s[HID0 + j]);
  #pragma unroll
  for (int u = 0; u < 8; ++u)
    for (int c = 0; c < 3; ++c)
      F1[(n * 8 + u) * 3 + c] = O1[(n * 8 + u) * 3 + c] * g[u];
  #pragma unroll
  for (int u = 0; u < 4; ++u)
    for (int c = 0; c < 5; ++c)
      F2[(n * 4 + u) * 5 + c] = O2[(n * 4 + u) * 5 + c] * g[8 + u];
}

__global__ void k_rv(const float* __restrict__ r1, const float* __restrict__ r2,
                     float* __restrict__ rv)
{
  int u = threadIdx.x;
  if (u >= 32) return;
  float a = 0.f;
  #pragma unroll
  for (int j = 0; j < 16; ++j) a += r1[u * 16 + j] * r2[j * 2];
  rv[u] = a;
}

__global__ void k_readout(const float* __restrict__ F0, const float* __restrict__ rv,
                          const int* __restrict__ gidx, float* __restrict__ out, int N)
{
  long n = (long)blockIdx.x * blockDim.x + threadIdx.x;
  if (n >= N) return;
  float a = 0.f;
  #pragma unroll
  for (int u = 0; u < 32; ++u) a += F0[n * 32 + u] * rv[u];
  atomicAdd(&out[gidx[n]], a);
}

// ---------------------------------------------------------------------------
// Host orchestration
// ---------------------------------------------------------------------------
static inline void gemmF(hipStream_t st, const float* A, const float* B, float* C,
                         int M, int Nn, int K, int ars, int brs, int crs,
                         float alpha, int act, int acc)
{
  dim3 g((M + 15) / 16, (Nn + 63) / 64);
  if (Nn % 64 == 0)
    k_gemm_fast<4,0><<<g, 32, 0, st>>>(A, B, C, M, K, ars, brs, crs, alpha, act, acc);
  else if (Nn == 32)
    k_gemm_fast<2,0><<<g, 32, 0, st>>>(A, B, C, M, K, ars, brs, crs, alpha, act, acc);
  else if (Nn == 44)
    k_gemm_fast<2,12><<<g, 32, 0, st>>>(A, B, C, M, K, ars, brs, crs, alpha, act, acc);
  else {
    dim3 gg((M + 15) / 16, (Nn + 15) / 16);
    k_gemm_gen<<<gg, 32, 0, st>>>(A, B, C, M, Nn, K, ars, 1, brs, 1, crs, 1,
                                  alpha, act, acc);
  }
}
static inline void gemmG(hipStream_t st, const float* A, const float* B, float* C,
                         int M, int Nn, int K,
                         int ars, int acs, int brs, int bcs, int crs, int ccs,
                         float alpha, int act, int acc)
{
  dim3 g((M + 15) / 16, (Nn + 15) / 16);
  k_gemm_gen<<<g, 32, 0, st>>>(A, B, C, M, Nn, K, ars, acs, brs, bcs, crs, ccs,
                               alpha, act, acc);
}

extern "C" void kernel_launch(void* const* d_in, const int* in_sizes, int n_in,
                              void* d_out, int out_size, void* d_ws, size_t ws_size,
                              hipStream_t stream)
{
  (void)n_in; (void)ws_size;
  const float* Rij     = (const float*)d_in[0];
  const int*   species = (const int*)d_in[1];
  const int*   snd     = (const int*)d_in[2];
  const int*   rcv     = (const int*)d_in[3];
  const int*   gidx    = (const int*)d_in[4];
  // d_in[5] = n_graphs scalar (compile-time 64 here)
  int E = in_sizes[0] / 3;
  int N = in_sizes[1];

  // Param leaves in JAX pytree order (sorted dict keys):
  // embed; per layer: out[0],out[1],out[2], skip[0](,skip[1],skip[2]),
  //                   up[0](,up[1],up[2]), w1, w2, w3; then r1, r2.
  int p = 6;
  const float* embed = (const float*)d_in[p++];
  const float *Lout0[4], *Lout1[4], *Lout2[4], *Lsk0[4], *Lsk1[4], *Lsk2[4];
  const float *Lup0[4], *Lup1[4], *Lup2[4], *Lw1[4], *Lw2[4], *Lw3[4];
  for (int i = 0; i < 4; ++i) {
    Lout0[i] = (const float*)d_in[p++];
    Lout1[i] = (const float*)d_in[p++];
    Lout2[i] = (const float*)d_in[p++];
    Lsk0[i]  = (const float*)d_in[p++];
    if (i > 0) { Lsk1[i] = (const float*)d_in[p++]; Lsk2[i] = (const float*)d_in[p++]; }
    else       { Lsk1[i] = nullptr; Lsk2[i] = nullptr; }
    Lup0[i]  = (const float*)d_in[p++];
    if (i > 0) { Lup1[i] = (const float*)d_in[p++]; Lup2[i] = (const float*)d_in[p++]; }
    else       { Lup1[i] = nullptr; Lup2[i] = nullptr; }
    Lw1[i] = (const float*)d_in[p++];
    Lw2[i] = (const float*)d_in[p++];
    Lw3[i] = (const float*)d_in[p++];
  }
  const float* r1 = (const float*)d_in[p++];
  const float* r2 = (const float*)d_in[p++];

  // Workspace carve-up (floats). All region starts are 16B-aligned.
  float* ws = (float*)d_ws;
  size_t o = 0;
  float* SH  = ws + o; o += (size_t)E * 9;
  float* RAD = ws + o; o += (size_t)E * 8;
  float* H1  = ws + o; o += (size_t)E * 64;
  float* H2  = ws + o; o += (size_t)E * 64;
  float* F0  = ws + o; o += (size_t)N * 128;
  float* F1  = ws + o; o += (size_t)N * 24;
  float* F2  = ws + o; o += (size_t)N * 20;
  float* X0  = ws + o; o += (size_t)N * 128;
  float* X1  = ws + o; o += (size_t)N * 24;
  float* X2  = ws + o; o += (size_t)N * 20;
  float* A0  = ws + o; o += (size_t)N * 128;
  float* A1  = ws + o; o += (size_t)N * 384;
  float* A2  = ws + o; o += (size_t)N * 640;
  float* O0  = ws + o; o += (size_t)N * 44;
  float* O1  = ws + o; o += (size_t)N * 24;
  float* O2  = ws + o; o += (size_t)N * 20;
  float* CGT = ws + o; o += 384;
  float* RV  = ws + o; o += 32;

  k_init_cg<<<1, 1, 0, stream>>>(CGT);
  k_edge_geom<<<(E + 255) / 256, 256, 0, stream>>>(Rij, SH, RAD, E);
  k_embed<<<(int)(((size_t)N * 128 + 255) / 256), 256, 0, stream>>>(embed, species, F0, N);

  for (int i = 0; i < 4; ++i) {
    int first = (i == 0);
    int mi0 = first ? 128 : 32;
    int nW  = first ? 384 : 144;
    int ch0 = first ? 128 : 44, ch1 = first ? 128 : 52, ch2 = first ? 128 : 48;

    // Channel up-mix (einsum num,uv->nvm) via WMMA GEMMs.
    if (first) {
      gemmF(stream, F0, Lup0[0], X0, N, 128, 128, 128, 128, 128, 1.f, 0, 0);
    } else {
      gemmF(stream, F0, Lup0[i], X0, N, 32, 32, 32, 32, 32, 1.f, 0, 0);
      for (int c = 0; c < 3; ++c)
        gemmG(stream, F1 + c, Lup1[i], X1 + c, N, 8, 8, 24, 3, 8, 1, 24, 3, 1.f, 0, 0);
      for (int c = 0; c < 5; ++c)
        gemmG(stream, F2 + c, Lup2[i], X2 + c, N, 4, 4, 20, 5, 4, 1, 20, 5, 1.f, 0, 0);
    }

    // Radial MLP stages 1-2 (fused SiLU).
    gemmF(stream, RAD, Lw1[i], H1, E, 64, 8,  8,  64, 64, 1.f, 1, 0);
    gemmF(stream, H1,  Lw2[i], H2, E, 64, 64, 64, 64, 64, 1.f, 1, 0);

    hipMemsetAsync(A0, 0, (size_t)N * ch0 * 1 * sizeof(float), stream);
    hipMemsetAsync(A1, 0, (size_t)N * ch1 * 3 * sizeof(float), stream);
    hipMemsetAsync(A2, 0, (size_t)N * ch2 * 5 * sizeof(float), stream);

    // Fused w3-GEMM + message + scatter.
    k_edge_msg<<<(E + 127) / 128, 256, 0, stream>>>(
        H2, Lw3[i], SH, X0, X1, X2, snd, rcv, CGT, A0, A1, A2,
        E, nW, first, mi0, 8, 4, ch0, ch1, ch2);

    // Skip connections initialize the outputs...
    k_skip<<<N, 64, 0, stream>>>(F0, Lsk0[i], species, O0, N, mi0, 44, 1);
    if (!first) {
      k_skip<<<N, 64, 0, stream>>>(F1, Lsk1[i], species, O1, N, 8, 8, 3);
      k_skip<<<N, 64, 0, stream>>>(F2, Lsk2[i], species, O2, N, 4, 4, 5);
    }
    // ...then output projections accumulate (alpha = 1/sqrt(AVG_NEIGH) = 0.25).
    gemmF(stream, A0, Lout0[i], O0, N, 44, ch0, ch0, 44, 44, 0.25f, 0, 1);
    for (int c = 0; c < 3; ++c)
      gemmG(stream, A1 + c, Lout1[i], O1 + c, N, 8, ch1, ch1 * 3, 3, 8, 1, 24, 3,
            0.25f, 0, first ? 0 : 1);
    for (int c = 0; c < 5; ++c)
      gemmG(stream, A2 + c, Lout2[i], O2 + c, N, 4, ch2, ch2 * 5, 5, 4, 1, 20, 5,
            0.25f, 0, first ? 0 : 1);

    k_gate<<<(N + 255) / 256, 256, 0, stream>>>(O0, O1, O2, F0, F1, F2, N);
  }

  hipMemsetAsync(d_out, 0, (size_t)out_size * sizeof(float), stream);
  k_rv<<<1, 32, 0, stream>>>(r1, r2, RV);
  k_readout<<<(N + 255) / 256, 256, 0, stream>>>(F0, RV, gidx, (float*)d_out, N);
}
